// DPFASTLSTM_40261023432953
// MI455X (gfx1250) — matrix operand order repaired and checked by reading, not compiled, not run on hardware
//
#include <hip/hip_runtime.h>
#include <hip/hip_bf16.h>

#define B_ 64
#define L_ 512
#define I_ 512
#define H_ 768
#define G_ 3072   // 4*H

typedef __attribute__((ext_vector_type(16))) __bf16 v16bf;
typedef __attribute__((ext_vector_type(8)))  float  v8f;

union ABfrag { v16bf v; uint4 u[2]; };

__device__ __forceinline__ float sigmoidf_(float x) {
  return 1.0f / (1.0f + __expf(-x));
}

// ---------------------------------------------------------------- utilities
__global__ void bd_zero_sync(unsigned* s) {
  if (threadIdx.x < 8) s[threadIdx.x] = 0u;
}

__global__ void bd_cvt(const float* __restrict__ s, __hip_bfloat16* __restrict__ d, int n) {
  for (int i = blockIdx.x * blockDim.x + threadIdx.x; i < n; i += gridDim.x * blockDim.x)
    d[i] = __float2bfloat16(s[i]);
}

// ---------------------------------------------------------------- phase 1:
// xg[dir][t][b][n] = x[b][l_eff][:] @ W_ih[dir][n][:] + bias[dir][n]   (bf16 out)
// 8 waves/block, each wave owns a 4Mx4N register tile (64x64), block tile 128x256.
__global__ void __launch_bounds__(256)
bd_ingemm(const __hip_bfloat16* __restrict__ xbf,
          const __hip_bfloat16* __restrict__ wih,
          const float* __restrict__ bias_f,
          const float* __restrict__ bias_r,
          __hip_bfloat16* __restrict__ xg) {
  const int lane = threadIdx.x & 31, wave = threadIdx.x >> 5;
  const int lm = lane & 15, lh = lane >> 4;
  const int dir = blockIdx.z;
  const int m0 = blockIdx.x * 128 + (wave >> 2) * 64;   // wave M base (4 tiles)
  const int n0 = blockIdx.y * 256 + (wave & 3) * 64;    // wave N base (4 tiles)

  const float* bias = dir ? bias_r : bias_f;
  const __hip_bfloat16* W = wih + (size_t)dir * G_ * I_;

  // A-matrix source rows for this lane, one per M tile (fixed across K)
  const __hip_bfloat16* arow[4];
#pragma unroll
  for (int mt = 0; mt < 4; ++mt) {
    const int m  = m0 + mt * 16 + lm;
    const int b  = m & (B_ - 1);
    const int ts = m >> 6;
    const int l_eff = dir ? (L_ - 1 - ts) : ts;
    arow[mt] = xbf + ((size_t)b * L_ + l_eff) * I_;
  }

  v8f acc[4][4];
  int ncol[4];
#pragma unroll
  for (int nt = 0; nt < 4; ++nt) {
    ncol[nt] = n0 + nt * 16 + lm;
    const float bv = bias[ncol[nt]];
#pragma unroll
    for (int mt = 0; mt < 4; ++mt)
#pragma unroll
      for (int r = 0; r < 8; ++r) acc[mt][nt][r] = bv;
  }

  for (int k0 = 0; k0 < I_; k0 += 32) {
    ABfrag a[4];
#pragma unroll
    for (int mt = 0; mt < 4; ++mt) {
      a[mt].u[0] = *(const uint4*)(arow[mt] + k0 + lh * 16);
      a[mt].u[1] = *(const uint4*)(arow[mt] + k0 + 8 + lh * 16);
    }
#pragma unroll
    for (int nt = 0; nt < 4; ++nt) {
      ABfrag bb;
      const __hip_bfloat16* brow = W + (size_t)ncol[nt] * I_ + k0 + lh * 16;
      bb.u[0] = ((const uint4*)brow)[0];
      bb.u[1] = ((const uint4*)brow)[1];
#pragma unroll
      for (int mt = 0; mt < 4; ++mt)
        acc[mt][nt] = __builtin_amdgcn_wmma_f32_16x16x32_bf16(
            false, a[mt].v, false, bb.v, (short)0, acc[mt][nt], false, false);
    }
  }

#pragma unroll
  for (int mt = 0; mt < 4; ++mt)
#pragma unroll
    for (int nt = 0; nt < 4; ++nt)
#pragma unroll
      for (int r = 0; r < 8; ++r) {
        const int mr = m0 + mt * 16 + r + lh * 8;
        const int b2 = mr & (B_ - 1), ts2 = mr >> 6;
        xg[(((size_t)dir * L_ + ts2) * B_ + b2) * G_ + ncol[nt]] =
            __float2bfloat16(acc[mt][nt][r]);
      }
}

// ---------------------------------------------------------------- phase 2:
// per-direction barrier across 16 WGs: cluster barrier fast path, atomic fallback
__device__ __forceinline__ void bd_dir_barrier(unsigned* ctr, unsigned target) {
  __syncthreads();
  __threadfence();                                  // release h writes to agent scope
  if (__builtin_amdgcn_cluster_id_x() != 0) {
    if ((threadIdx.x >> 5) == 0)                    // one wave per WG signals+waits
      __builtin_amdgcn_s_cluster_barrier();         // s_barrier_signal -3 / s_barrier_wait -3
  } else {
    if (threadIdx.x == 0) {
      __hip_atomic_fetch_add(ctr, 1u, __ATOMIC_RELEASE, __HIP_MEMORY_SCOPE_AGENT);
      while (__hip_atomic_load(ctr, __ATOMIC_ACQUIRE, __HIP_MEMORY_SCOPE_AGENT) < target)
        __builtin_amdgcn_s_sleep(2);
    }
  }
  __syncthreads();
  __threadfence();                                  // acquire side
}

// Recurrent scan. grid = (16 WGs, 2 dirs), block = 384 (12 waves).
// Each WG owns 48 h-columns: LDS holds its W_hh slice [192 gate-rows][768 K] (288KB,
// staged via GLOBAL_LOAD_ASYNC_TO_LDS) plus a bf16 gate staging buffer [192][64] (24KB).
// Waves 0-3 each compute one full gate as a 4Mx3N WMMA register tile; c state in regs.
__global__ void __launch_bounds__(384)
bd_scan(const __hip_bfloat16* __restrict__ whh,
        const __hip_bfloat16* __restrict__ xg,
        const float* __restrict__ h0,
        const float* __restrict__ c0,
        __hip_bfloat16* __restrict__ hbf,   // [dir][buf][64][768] bf16, double buffered
        float* __restrict__ out,
        unsigned* __restrict__ sync) {
  extern __shared__ unsigned char smem[];
  __hip_bfloat16* sW = (__hip_bfloat16*)smem;                    // [192][768]
  __hip_bfloat16* sG = (__hip_bfloat16*)(smem + 192 * H_ * 2);   // [192][64]

  const int tid = threadIdx.x, lane = tid & 31, wave = tid >> 5;
  const int lm = lane & 15, lh = lane >> 4;
  const int wg  = blockIdx.x;          // 0..15
  const int dir = blockIdx.y;          // 0..1
  const int ccol = wg * 48;            // h-column base owned by this WG

  // ---- stage W_hh slice into LDS with async copies (ASYNCcnt-tracked, VGPR-bypass)
  const __hip_bfloat16* Wd = whh + (size_t)dir * G_ * H_;
  {
    const unsigned ldsbase = (unsigned)(uintptr_t)sW;   // low 32 bits = LDS offset
    for (int idx = tid; idx < 192 * (H_ / 8); idx += 384) {
      const int nl = idx / (H_ / 8);
      const int ck = idx % (H_ / 8);
      const int ng = (nl / 48) * H_ + ccol + (nl % 48);
      const unsigned lds_addr = ldsbase + (unsigned)(nl * H_ + ck * 8) * 2u;
      const unsigned long long ga =
          (unsigned long long)(uintptr_t)(Wd + (size_t)ng * H_ + ck * 8);
      asm volatile("global_load_async_to_lds_b128 %0, %1, off"
                   :: "v"(lds_addr), "v"(ga) : "memory");
    }
    asm volatile("s_wait_asynccnt 0" ::: "memory");
  }

  // ---- init h (buffer 0) and c registers for our 48 columns
  const int jl = tid % 48;             // column within chunk
  const int mg = tid / 48;             // 0..7 (each handles 8 batch rows)
  float creg[8];
  {
    const float* h0d = h0 + (size_t)dir * B_ * H_;
    const float* c0d = c0 + (size_t)dir * B_ * H_;
#pragma unroll
    for (int q = 0; q < 8; ++q) {
      const int mm = mg * 8 + q;
      creg[q] = c0d[(size_t)mm * H_ + ccol + jl];
      hbf[(((size_t)dir * 2 + 0) * B_ + mm) * H_ + ccol + jl] =
          __float2bfloat16(h0d[(size_t)mm * H_ + ccol + jl]);
    }
  }
  bd_dir_barrier(sync + dir, 16u);

  // GEMM waves: wave w (0..3) owns gate w -> LDS rows [w*48, w*48+48)
  const bool gemmw = (wave < 4);
  const int nl0 = wave * 48;

  for (int t = 0; t < L_; ++t) {
    const __hip_bfloat16* xgt = xg + (((size_t)dir * L_ + t) * B_) * G_;
    const __hip_bfloat16* hread = hbf + (((size_t)dir * 2 + (t & 1)) * B_) * H_;

    if (gemmw) {
      // accumulators seeded from xg_t (carries x-proj + bias)
      v8f acc[4][3];
#pragma unroll
      for (int mt = 0; mt < 4; ++mt)
#pragma unroll
        for (int nt = 0; nt < 3; ++nt)
#pragma unroll
          for (int r = 0; r < 8; ++r)
            acc[mt][nt][r] = __bfloat162float(
                xgt[(size_t)(mt * 16 + r + lh * 8) * G_ +
                    wave * H_ + ccol + nt * 16 + lm]);

      for (int k0 = 0; k0 < H_; k0 += 32) {
        ABfrag a[4];
#pragma unroll
        for (int mt = 0; mt < 4; ++mt) {
          const __hip_bfloat16* ap = hread + (size_t)(mt * 16 + lm) * H_ + k0 + lh * 16;
          a[mt].u[0] = *(const uint4*)ap;
          a[mt].u[1] = *(const uint4*)(ap + 8);
        }
#pragma unroll
        for (int nt = 0; nt < 3; ++nt) {
          ABfrag bf;
          const __hip_bfloat16* bp =
              sW + (size_t)(nl0 + nt * 16 + lm) * H_ + k0 + lh * 16;
          bf.u[0] = ((const uint4*)bp)[0];
          bf.u[1] = ((const uint4*)bp)[1];
#pragma unroll
          for (int mt = 0; mt < 4; ++mt)
            acc[mt][nt] = __builtin_amdgcn_wmma_f32_16x16x32_bf16(
                false, a[mt].v, false, bf.v, (short)0, acc[mt][nt], false, false);
        }
      }

      // stage raw gate pre-activations to LDS (bf16)
#pragma unroll
      for (int mt = 0; mt < 4; ++mt)
#pragma unroll
        for (int nt = 0; nt < 3; ++nt)
#pragma unroll
          for (int r = 0; r < 8; ++r)
            sG[(size_t)(nl0 + nt * 16 + lm) * B_ + mt * 16 + r + lh * 8] =
                __float2bfloat16(acc[mt][nt][r]);
    }
    __syncthreads();

    // prefetch next step's xg tile while we do the pointwise update
    if (t + 1 < L_ && tid < 256) {
      const int pm = tid & 63, pg = tid >> 6;
      __builtin_prefetch(
          xg + (((size_t)dir * L_ + (t + 1)) * B_ + pm) * G_ + pg * H_ + ccol, 0, 0);
    }

    // pointwise LSTM cell update for our 48 columns x 64 batches
    __hip_bfloat16* hwrite = hbf + (((size_t)dir * 2 + ((t + 1) & 1)) * B_) * H_;
    const int l_eff = dir ? (L_ - 1 - t) : t;
#pragma unroll
    for (int q = 0; q < 8; ++q) {
      const int mm = mg * 8 + q;
      const float iv = sigmoidf_(__bfloat162float(sG[(size_t)(0 * 48 + jl) * B_ + mm]));
      const float fv = sigmoidf_(__bfloat162float(sG[(size_t)(1 * 48 + jl) * B_ + mm]));
      const float gv = tanhf    (__bfloat162float(sG[(size_t)(2 * 48 + jl) * B_ + mm]));
      const float ov = sigmoidf_(__bfloat162float(sG[(size_t)(3 * 48 + jl) * B_ + mm]));
      const float cn = fv * creg[q] + iv * gv;
      creg[q] = cn;
      const float hn = ov * tanhf(cn);
      hwrite[(size_t)mm * H_ + ccol + jl] = __float2bfloat16(hn);
      // h_all: (B, L, 2H)
      out[((size_t)mm * L_ + l_eff) * (2 * H_) + (size_t)dir * H_ + ccol + jl] = hn;
      if (t == L_ - 1) {
        const size_t base = (size_t)B_ * L_ * 2 * H_;
        out[base + ((size_t)dir * B_ + mm) * H_ + ccol + jl] = hn;                       // h_n
        out[base + (size_t)2 * B_ * H_ + ((size_t)dir * B_ + mm) * H_ + ccol + jl] = cn; // c_n
      }
    }

    if (t + 1 < L_) bd_dir_barrier(sync + dir, 16u * (unsigned)(t + 2));
  }
}

// ---------------------------------------------------------------- launcher
extern "C" void kernel_launch(void* const* d_in, const int* in_sizes, int n_in,
                              void* d_out, int out_size, void* d_ws, size_t ws_size,
                              hipStream_t stream) {
  const float* x     = (const float*)d_in[0];
  const float* wih   = (const float*)d_in[1];
  const float* whh   = (const float*)d_in[2];
  const float* bias  = (const float*)d_in[3];
  const float* wihr  = (const float*)d_in[4];
  const float* whhr  = (const float*)d_in[5];
  const float* biasr = (const float*)d_in[6];
  const float* h0    = (const float*)d_in[7];
  const float* c0    = (const float*)d_in[8];
  float* out = (float*)d_out;

  char* ws = (char*)d_ws;
  unsigned*       sync = (unsigned*)ws;                         // 256 B
  __hip_bfloat16* xbf  = (__hip_bfloat16*)(ws + 256);           // 32 MB
  __hip_bfloat16* wihb = xbf  + (size_t)B_ * L_ * I_;           // 6 MB (2 dirs)
  __hip_bfloat16* whhb = wihb + (size_t)2 * G_ * I_;            // 9 MB (2 dirs)
  __hip_bfloat16* hbf  = whhb + (size_t)2 * G_ * H_;            // 384 KB (2 dirs x 2 bufs)
  __hip_bfloat16* xgb  = hbf  + (size_t)2 * 2 * B_ * H_;        // ~403 MB

  bd_zero_sync<<<1, 64, 0, stream>>>(sync);
  bd_cvt<<<4096, 256, 0, stream>>>(x,    xbf,  B_ * L_ * I_);
  bd_cvt<<<2048, 256, 0, stream>>>(wih,  wihb,            G_ * I_);
  bd_cvt<<<2048, 256, 0, stream>>>(wihr, wihb + (size_t)G_ * I_, G_ * I_);
  bd_cvt<<<2048, 256, 0, stream>>>(whh,  whhb,            G_ * H_);
  bd_cvt<<<2048, 256, 0, stream>>>(whhr, whhb + (size_t)G_ * H_, G_ * H_);

  bd_ingemm<<<dim3(256, 12, 2), 256, 0, stream>>>(xbf, wihb, bias, biasr, xgb);

  const size_t lds_bytes = (size_t)192 * H_ * 2 + (size_t)192 * B_ * 2;  // 312 KB
  bd_scan<<<dim3(16, 2, 1), 384, lds_bytes, stream>>>(whhb, xgb, h0, c0, hbf, out, sync);
}
